// Block_16303695856254
// MI455X (gfx1250) — matrix operand order repaired and checked by reading, not compiled, not run on hardware
//
#include <hip/hip_runtime.h>
#include <hip/hip_bf16.h>
#include <math.h>

// ---------------- problem constants ----------------
#define T_SEQ   4096
#define C_DIM   2048
#define NHEADS  16
#define NGROUPS 4
#define HS      128
#define INTER   5632
#define QKV_OUT 3072   // (16 + 2*4) * 128
#define RMS_EPS 1e-5f

typedef __attribute__((ext_vector_type(16))) __bf16 v16bf;
typedef __attribute__((ext_vector_type(8)))  float  v8f;
typedef __attribute__((ext_vector_type(4)))  int    v4i;

union FragBF { v16bf v; uint4 q[2]; unsigned short us[16]; };

static __device__ __forceinline__ unsigned short f2bf_bits(float f) {
    union { __bf16 b; unsigned short u; } cv; cv.b = (__bf16)f; return cv.u;
}

// ---------------- gfx1250 async global->LDS copy (ASYNCcnt path) ----------------
#if defined(__HIP_DEVICE_COMPILE__) && __has_builtin(__builtin_amdgcn_global_load_async_to_lds_b128)
#define HAVE_ASYNC_LDS 1
#else
#define HAVE_ASYNC_LDS 0
#endif

static __device__ __forceinline__ void async_cp16(const __bf16* __restrict__ g,
                                                  unsigned short* __restrict__ l) {
#if HAVE_ASYNC_LDS
    __builtin_amdgcn_global_load_async_to_lds_b128(
        (__attribute__((address_space(1))) v4i*)g,
        (__attribute__((address_space(3))) v4i*)l, 0, 0);
#else
    *(uint4*)l = *(const uint4*)g;     // fallback: sync copy through VGPRs
#endif
}

#if defined(__HIP_DEVICE_COMPILE__) && __has_builtin(__builtin_amdgcn_s_wait_asynccnt)
#define WAIT_ASYNC(n) __builtin_amdgcn_s_wait_asynccnt(n)
#elif HAVE_ASYNC_LDS
#define WAIT_ASYNC(n) asm volatile("s_wait_asynccnt %0" :: "i"(n) : "memory")
#else
#define WAIT_ASYNC(n) ((void)0)
#endif

// ---------------- elementwise kernels ----------------
__global__ void cast_f32_to_bf16(const float* __restrict__ in,
                                 __bf16* __restrict__ out, long n) {
    long i = (long)blockIdx.x * blockDim.x + threadIdx.x;
    long stride = (long)gridDim.x * blockDim.x;
    for (; i < n; i += stride) out[i] = (__bf16)in[i];
}

__global__ __launch_bounds__(256)
void rmsnorm_bf16(const float* __restrict__ x, const float* __restrict__ w,
                  __bf16* __restrict__ out) {
    int t = blockIdx.x;
    int tid = threadIdx.x;
    const float4* xr = (const float4*)(x + (long)t * C_DIM);
    const float4* wr = (const float4*)w;
    __shared__ float red[8];
    float4 v0 = xr[tid];
    float4 v1 = xr[tid + 256];
    float ss = v0.x*v0.x + v0.y*v0.y + v0.z*v0.z + v0.w*v0.w
             + v1.x*v1.x + v1.y*v1.y + v1.z*v1.z + v1.w*v1.w;
    for (int m = 1; m < 32; m <<= 1) ss += __shfl_xor(ss, m, 32);
    if ((tid & 31) == 0) red[tid >> 5] = ss;
    __syncthreads();
    float tot = red[0]+red[1]+red[2]+red[3]+red[4]+red[5]+red[6]+red[7];
    float rs = rsqrtf(tot / (float)C_DIM + RMS_EPS);
    __bf16* o = out + (long)t * C_DIM;
    float4 w0 = wr[tid], w1 = wr[tid + 256];
    int b0 = tid * 4, b1 = (tid + 256) * 4;
    o[b0+0] = (__bf16)(v0.x * rs * w0.x);  o[b0+1] = (__bf16)(v0.y * rs * w0.y);
    o[b0+2] = (__bf16)(v0.z * rs * w0.z);  o[b0+3] = (__bf16)(v0.w * rs * w0.w);
    o[b1+0] = (__bf16)(v1.x * rs * w1.x);  o[b1+1] = (__bf16)(v1.y * rs * w1.y);
    o[b1+2] = (__bf16)(v1.z * rs * w1.z);  o[b1+3] = (__bf16)(v1.w * rs * w1.w);
}

// qkv fp32 [T, 3072] -> rope(q,k) + split into bf16 q[h][T][128], k[g][T][128], v[g][T][128]
__global__ void rope_split(const float* __restrict__ qkv,
                           const float* __restrict__ cosb, const float* __restrict__ sinb,
                           __bf16* __restrict__ q, __bf16* __restrict__ k,
                           __bf16* __restrict__ v) {
    int t = blockIdx.x;
    int slot = blockIdx.y;           // 0..23 = g*6 + s
    int g = slot / 6, s = slot % 6;
    int d = threadIdx.x;             // 0..127
    long base = (long)t * QKV_OUT + (long)slot * HS;
    float xv = qkv[base + d];
    float outv;
    if (s < 5) { // rope on q and k
        float c  = cosb[(long)t * HS + d];
        float sn = sinb[(long)t * HS + d];
        float pair = (d < HS/2) ? -qkv[base + d + HS/2] : qkv[base + d - HS/2];
        outv = xv * c + pair * sn;
    } else {
        outv = xv;
    }
    if (s < 4) {
        int h = g * 4 + s;
        q[((long)h * T_SEQ + t) * HS + d] = (__bf16)outv;
    } else if (s == 4) {
        k[((long)g * T_SEQ + t) * HS + d] = (__bf16)outv;
    } else {
        v[((long)g * T_SEQ + t) * HS + d] = (__bf16)outv;
    }
}

// ---------------- bf16 GEMM: acc[M,N] = A[M,K] * B[N,K]^T, fused epilogues ----------
// Block tile 128x256, 8 waves (2x4), wave tile 64x64. BK=32, double-buffered LDS,
// staged with GLOBAL_LOAD_ASYNC_TO_LDS_B128 (6 x b128 per thread per tile).
// mode 0: Cout = acc                      (f32)
// mode 1: Cout = acc + aux                (f32 residual add)
// mode 2: outBF = bf16(silu(aux) * acc)   (fused SwiGLU epilogue; acc = "up")
#define BM 128
#define BN 256
#define BK 32
#define LDH 40   // LDS row stride in halves (80B, 16B aligned, staggers banks)

static __device__ __forceinline__ void gemm_stage(
    const __bf16* __restrict__ A, const __bf16* __restrict__ Bw,
    unsigned short* __restrict__ As, unsigned short* __restrict__ Bs,
    int K, int tileM, int tileN, int k0, int tid) {
    for (int i = 0; i < 2; ++i) {              // A: 128x32 halves = 512 b128 chunks
        int c = tid * 2 + i;
        int row = c >> 2, off = (c & 3) * 8;
        async_cp16(&A[(long)(tileM + row) * K + k0 + off], &As[row * LDH + off]);
    }
    for (int i = 0; i < 4; ++i) {              // B: 256x32 halves = 1024 b128 chunks
        int c = tid * 4 + i;
        int row = c >> 2, off = (c & 3) * 8;
        async_cp16(&Bw[(long)(tileN + row) * K + k0 + off], &Bs[row * LDH + off]);
    }
}

__global__ __launch_bounds__(256)
void gemm_bf16(const __bf16* __restrict__ A, const __bf16* __restrict__ Bw,
               const float* __restrict__ aux, float* __restrict__ Cout,
               __bf16* __restrict__ outBF,
               int M, int N, int K, int mode) {
    __shared__ __align__(16) unsigned short As[2][BM * LDH];
    __shared__ __align__(16) unsigned short Bs[2][BN * LDH];
    int tid = threadIdx.x;
    int lane = tid & 31, wid = tid >> 5;
    int wm = wid >> 2, wn = wid & 3;        // 2 x 4 wave grid
    int laneN = lane & 15, hi = lane >> 4;  // lane half (ISA A/B fragment layouts)
    int tileM = blockIdx.y * BM, tileN = blockIdx.x * BN;

    v8f acc[4][4];
    v8f vzero = {0.f,0.f,0.f,0.f,0.f,0.f,0.f,0.f};
    for (int r = 0; r < 4; ++r) for (int t2 = 0; t2 < 4; ++t2) acc[r][t2] = vzero;

    int nk = K / BK;
    gemm_stage(A, Bw, As[0], Bs[0], K, tileM, tileN, 0, tid);   // prologue

    for (int ib = 0; ib < nk; ++ib) {
        if (ib + 1 < nk) {
            gemm_stage(A, Bw, As[(ib + 1) & 1], Bs[(ib + 1) & 1],
                       K, tileM, tileN, (ib + 1) * BK, tid);
            WAIT_ASYNC(6);      // batch ib complete; batch ib+1 still in flight
        } else {
            WAIT_ASYNC(0);
        }
        __syncthreads();

        const unsigned short* Ab = As[ib & 1];
        const unsigned short* Bb = Bs[ib & 1];
        FragBF aF[4], bF[4];
        for (int r = 0; r < 4; ++r) {
            // A 16x32 bf16 layout: lane=M, K chunks 0-7/16-23 (lanes<16) vs 8-15/24-31
            int row = wm * 64 + r * 16 + laneN;
            aF[r].q[0] = *(const uint4*)&Ab[row * LDH + hi * 8];
            aF[r].q[1] = *(const uint4*)&Ab[row * LDH + hi * 8 + 16];
        }
        for (int t2 = 0; t2 < 4; ++t2) {
            // B 32x16 bf16 layout: lane=N, K chunks 0-7/16-23 (lanes<16) vs 8-15/24-31
            int n = wn * 64 + t2 * 16 + laneN;
            bF[t2].q[0] = *(const uint4*)&Bb[n * LDH + hi * 8];
            bF[t2].q[1] = *(const uint4*)&Bb[n * LDH + hi * 8 + 16];
        }
        for (int r = 0; r < 4; ++r)
            for (int t2 = 0; t2 < 4; ++t2)
                acc[r][t2] = __builtin_amdgcn_wmma_f32_16x16x32_bf16(
                    false, aF[r].v, false, bF[t2].v, (short)0, acc[r][t2], false, false);
        __syncthreads();        // frees buffer (ib&1) for the stage at ib+1
    }

    // C layout: VGPR i -> rows i (lanes 0-15) / i+8 (lanes 16-31), N = laneN
    for (int r = 0; r < 4; ++r)
        for (int t2 = 0; t2 < 4; ++t2)
            for (int i = 0; i < 8; ++i) {
                int row = tileM + wm * 64 + r * 16 + hi * 8 + i;
                int col = tileN + wn * 64 + t2 * 16 + laneN;
                long idx = (long)row * N + col;
                float vv = acc[r][t2][i];
                if (mode == 2) {
                    float gv = aux[idx];
                    float sv = gv / (1.0f + __expf(-gv));
                    outBF[idx] = (__bf16)(sv * vv);
                } else if (mode == 1) {
                    Cout[idx] = vv + aux[idx];
                } else {
                    Cout[idx] = vv;
                }
            }
}

// ---------------- causal flash attention (GQA 16 heads / 4 kv groups) ----------------
__global__ __launch_bounds__(128)
void flash_attn(const __bf16* __restrict__ qb, const __bf16* __restrict__ kbuf,
                const __bf16* __restrict__ vbuf, __bf16* __restrict__ yb) {
    __shared__ __align__(16) unsigned short Kt[32 * 136];   // k rows (row-major)
    __shared__ __align__(16) unsigned short Vt[128 * 40];   // v transposed [d][j]
    __shared__ __align__(16) unsigned short Pt[4][16 * 40]; // per-wave P bounce
    int h = blockIdx.y;
    int g = h >> 2;
    int Q0 = blockIdx.x * 64;
    int tid = threadIdx.x, lane = tid & 31, w = tid >> 5;
    int laneN = lane & 15, hi = lane >> 4;
    int qrow = Q0 + w * 16 + laneN;

    // preload q A-fragments for all 4 K-steps over HS=128
    FragBF qF[4];
    for (int kk = 0; kk < 4; ++kk) {
        long base = ((long)h * T_SEQ + qrow) * HS + kk * 32 + hi * 8;
        qF[kk].q[0] = *(const uint4*)&qb[base];
        qF[kk].q[1] = *(const uint4*)&qb[base + 16];
    }

    v8f o[8];
    v8f vzero = {0.f,0.f,0.f,0.f,0.f,0.f,0.f,0.f};
    for (int dn = 0; dn < 8; ++dn) o[dn] = vzero;
    float m_run[8], l_run[8];
    for (int i = 0; i < 8; ++i) { m_run[i] = -INFINITY; l_run[i] = 0.f; }

    int rowlast = Q0 + w * 16 + 15;
    int nkb = (Q0 + 64) >> 5;
    const float scale = 0.08838834764831845f; // 1/sqrt(128)

    for (int b = 0; b < nkb; ++b) {
        int kb0 = b * 32;
        __syncthreads();                      // prior iteration's LDS reads done
        // stage K rows kb0..kb0+31 via async global->LDS (4 x b128 per thread)
        for (int i = 0; i < 4; ++i) {
            int c = tid * 4 + i;              // 0..511 chunks of 8 halves
            int row = c >> 4, off = (c & 15) * 8;
            async_cp16(&kbuf[((long)g * T_SEQ + kb0 + row) * HS + off],
                       &Kt[row * 136 + off]);
        }
        // stage V transposed: Vt[d][j]; pack two rows per b32 store
        for (int it = 0; it < 2; ++it) {
            int w2 = tid * 2 + it;            // 0..255: (row-pair, d-chunk)
            int jp = w2 >> 4, ch = w2 & 15;
            int j = jp * 2;
            const uint4 ra = *(const uint4*)&vbuf[((long)g * T_SEQ + kb0 + j)     * HS + ch * 8];
            const uint4 rb = *(const uint4*)&vbuf[((long)g * T_SEQ + kb0 + j + 1) * HS + ch * 8];
            unsigned int wa[4] = {ra.x, ra.y, ra.z, ra.w};
            unsigned int wb[4] = {rb.x, rb.y, rb.z, rb.w};
            for (int qq = 0; qq < 4; ++qq) {
                int d = ch * 8 + qq * 2;
                unsigned int p0 = (wa[qq] & 0xffffu) | (wb[qq] << 16);
                unsigned int p1 = (wa[qq] >> 16) | (wb[qq] & 0xffff0000u);
                *(unsigned int*)&Vt[d * 40 + j]       = p0;
                *(unsigned int*)&Vt[(d + 1) * 40 + j] = p1;
            }
        }
        WAIT_ASYNC(0);
        __syncthreads();

        if (kb0 <= rowlast) {               // wave-uniform: EXEC stays all-ones for WMMA
            v8f s[2]; s[0] = vzero; s[1] = vzero;
            for (int nh = 0; nh < 2; ++nh) {
                for (int kk = 0; kk < 4; ++kk) {
                    FragBF bF;
                    int n = nh * 16 + laneN;
                    int off = kk * 32 + hi * 8;
                    bF.q[0] = *(const uint4*)&Kt[n * 136 + off];
                    bF.q[1] = *(const uint4*)&Kt[n * 136 + off + 16];
                    s[nh] = __builtin_amdgcn_wmma_f32_16x16x32_bf16(
                        false, qF[kk].v, false, bF.v, (short)0, s[nh], false, false);
                }
            }
            // scale + causal mask + online softmax
            for (int i = 0; i < 8; ++i) {
                int row = Q0 + w * 16 + hi * 8 + i;
                float a0 = s[0][i] * scale;
                float a1 = s[1][i] * scale;
                if (kb0 + laneN      > row) a0 = -INFINITY;
                if (kb0 + 16 + laneN > row) a1 = -INFINITY;
                float mx = fmaxf(a0, a1);
                for (int msk = 1; msk < 16; msk <<= 1)
                    mx = fmaxf(mx, __shfl_xor(mx, msk, 32));
                float mnew = fmaxf(m_run[i], mx);
                float fct = __expf(m_run[i] - mnew);
                m_run[i] = mnew;
                float p0 = __expf(a0 - mnew);
                float p1 = __expf(a1 - mnew);
                float rs = p0 + p1;
                for (int msk = 1; msk < 16; msk <<= 1)
                    rs += __shfl_xor(rs, msk, 32);
                l_run[i] = l_run[i] * fct + rs;
                for (int dn = 0; dn < 8; ++dn) o[dn][i] *= fct;
                // P (C-layout) -> LDS so we can re-read it in A-fragment layout
                Pt[w][(hi * 8 + i) * 40 + laneN]      = f2bf_bits(p0);
                Pt[w][(hi * 8 + i) * 40 + 16 + laneN] = f2bf_bits(p1);
            }
            asm volatile("s_wait_dscnt 0" ::: "memory"); // same-wave cross-lane LDS RAW
            FragBF pF;
            pF.q[0] = *(const uint4*)&Pt[w][laneN * 40 + hi * 8];
            pF.q[1] = *(const uint4*)&Pt[w][laneN * 40 + hi * 8 + 16];
            for (int dn = 0; dn < 8; ++dn) {
                FragBF vF;
                int d = dn * 16 + laneN;
                vF.q[0] = *(const uint4*)&Vt[d * 40 + hi * 8];
                vF.q[1] = *(const uint4*)&Vt[d * 40 + hi * 8 + 16];
                o[dn] = __builtin_amdgcn_wmma_f32_16x16x32_bf16(
                    false, pF.v, false, vF.v, (short)0, o[dn], false, false);
            }
        }
    }
    // epilogue: normalize and scatter y[t][h*128+d] (bf16 for the proj GEMM)
    for (int dn = 0; dn < 8; ++dn)
        for (int i = 0; i < 8; ++i) {
            int row = Q0 + w * 16 + hi * 8 + i;
            int col = h * HS + dn * 16 + laneN;
            yb[(long)row * C_DIM + col] = (__bf16)(o[dn][i] / l_run[i]);
        }
}

// ---------------- host orchestration ----------------
extern "C" void kernel_launch(void* const* d_in, const int* in_sizes, int n_in,
                              void* d_out, int out_size, void* d_ws, size_t ws_size,
                              hipStream_t stream) {
    const float* x       = (const float*)d_in[0];
    const float* cosb    = (const float*)d_in[1];
    const float* sinb    = (const float*)d_in[2];
    const float* norm1_w = (const float*)d_in[3];
    const float* norm2_w = (const float*)d_in[4];
    const float* attn_w  = (const float*)d_in[5];
    const float* proj_w  = (const float*)d_in[6];
    const float* gate_w  = (const float*)d_in[7];
    const float* up_w    = (const float*)d_in[8];
    const float* down_w  = (const float*)d_in[9];
    float* out = (float*)d_out;

    char* ws = (char*)d_ws;
    size_t off = 0;
    auto alloc = [&](size_t bytes) { char* p = ws + off; off += (bytes + 255) & ~size_t(255); return p; };
    __bf16* wAttn = (__bf16*)alloc((size_t)QKV_OUT * C_DIM * 2);
    __bf16* wProj = (__bf16*)alloc((size_t)C_DIM * C_DIM * 2);
    __bf16* wGate = (__bf16*)alloc((size_t)INTER * C_DIM * 2);
    __bf16* wUp   = (__bf16*)alloc((size_t)INTER * C_DIM * 2);
    __bf16* wDown = (__bf16*)alloc((size_t)C_DIM * INTER * 2);
    __bf16* xn1   = (__bf16*)alloc((size_t)T_SEQ * C_DIM * 2);
    float*  qkv   = (float*) alloc((size_t)T_SEQ * QKV_OUT * 4);
    __bf16* qB    = (__bf16*)alloc((size_t)NHEADS * T_SEQ * HS * 2);
    __bf16* kB    = (__bf16*)alloc((size_t)NGROUPS * T_SEQ * HS * 2);
    __bf16* vB    = (__bf16*)alloc((size_t)NGROUPS * T_SEQ * HS * 2);
    __bf16* yB    = (__bf16*)alloc((size_t)T_SEQ * C_DIM * 2);
    float*  x2    = (float*) alloc((size_t)T_SEQ * C_DIM * 4);
    __bf16* xn2   = (__bf16*)alloc((size_t)T_SEQ * C_DIM * 2);
    float*  gateF = (float*) alloc((size_t)T_SEQ * INTER * 4);
    __bf16* hB    = (__bf16*)alloc((size_t)T_SEQ * INTER * 2);

    // 1. weights -> bf16
    cast_f32_to_bf16<<<2048, 256, 0, stream>>>(attn_w, wAttn, (long)QKV_OUT * C_DIM);
    cast_f32_to_bf16<<<2048, 256, 0, stream>>>(proj_w, wProj, (long)C_DIM * C_DIM);
    cast_f32_to_bf16<<<2048, 256, 0, stream>>>(gate_w, wGate, (long)INTER * C_DIM);
    cast_f32_to_bf16<<<2048, 256, 0, stream>>>(up_w,   wUp,   (long)INTER * C_DIM);
    cast_f32_to_bf16<<<2048, 256, 0, stream>>>(down_w, wDown, (long)C_DIM * INTER);

    // 2. xn1 = rmsnorm(x) * norm1_w  (bf16)
    rmsnorm_bf16<<<T_SEQ, 256, 0, stream>>>(x, norm1_w, xn1);

    // 3. qkv = xn1 @ attn_w^T  (fp32, M=4096 N=3072 K=2048)
    gemm_bf16<<<dim3(QKV_OUT / BN, T_SEQ / BM), 256, 0, stream>>>(
        xn1, wAttn, nullptr, qkv, nullptr, T_SEQ, QKV_OUT, C_DIM, 0);

    // 4. rope + split -> q/k/v bf16
    rope_split<<<dim3(T_SEQ, 24), HS, 0, stream>>>(qkv, cosb, sinb, qB, kB, vB);

    // 5. causal flash attention -> y bf16 [T, 2048]
    flash_attn<<<dim3(T_SEQ / 64, NHEADS), 128, 0, stream>>>(qB, kB, vB, yB);

    // 6. x2 = x + y @ proj_w^T  (fp32 residual fuse)
    gemm_bf16<<<dim3(C_DIM / BN, T_SEQ / BM), 256, 0, stream>>>(
        yB, wProj, x, x2, nullptr, T_SEQ, C_DIM, C_DIM, 1);

    // 7. xn2 = rmsnorm(x2) * norm2_w  (bf16)
    rmsnorm_bf16<<<T_SEQ, 256, 0, stream>>>(x2, norm2_w, xn2);

    // 8. gate = xn2 @ gate_w^T (fp32), then up-GEMM with fused SwiGLU epilogue:
    //    hB = bf16( silu(gate) * (xn2 @ up_w^T) )
    gemm_bf16<<<dim3(INTER / BN, T_SEQ / BM), 256, 0, stream>>>(
        xn2, wGate, nullptr, gateF, nullptr, T_SEQ, INTER, C_DIM, 0);
    gemm_bf16<<<dim3(INTER / BN, T_SEQ / BM), 256, 0, stream>>>(
        xn2, wUp, gateF, nullptr, hB, T_SEQ, INTER, C_DIM, 2);

    // 9. out = x2 + hB @ mlp_proj^T  (fp32 residual fuse, K=5632)
    gemm_bf16<<<dim3(C_DIM / BN, T_SEQ / BM), 256, 0, stream>>>(
        hB, wDown, x2, out, nullptr, T_SEQ, C_DIM, INTER, 1);
}
